// GraphSMOTE_26362509263555
// MI455X (gfx1250) — matrix operand order repaired and checked by reading, not compiled, not run on hardware
//
#include <hip/hip_runtime.h>

typedef __attribute__((ext_vector_type(16))) _Float16 v16h;
typedef __attribute__((ext_vector_type(8)))  _Float16 v8h;
typedef __attribute__((ext_vector_type(8)))  float    v8f;

// ---------------------------------------------------------------------------
// WMMA fragment layouts used here (contiguous k per lane half, wave32):
//  A 16x32 f16:  lane(m=l16, half): a[i]   = A[m][k0 + half*16 + i]      i<16
//  B 32x16 f16 (computing A @ W^T, B[k][n] = W[n][k]):
//                lane(n=l16, half): b[i]   = W[n][k0 + half*16 + i]
//  C/D 16x16 f32: lane(n=l16, half): c[v]  = C[half*8 + v][n]
// ---------------------------------------------------------------------------
__device__ __forceinline__ v16h load_a_frag_f32(const float* A, int lda, int row,
                                                int k0, int half) {
  const float* p = A + (long)row * lda + k0 + half * 16;
  v16h a;
#pragma unroll
  for (int i = 0; i < 8; ++i) a[i] = (_Float16)p[i];
#pragma unroll
  for (int i = 0; i < 8; ++i) a[8 + i] = (_Float16)p[8 + i];
  return a;
}

__device__ __forceinline__ v16h load_b_frag_f32(const float* Brow, int k0, int half) {
  const float* p = Brow + k0 + half * 16;
  v16h b;
#pragma unroll
  for (int i = 0; i < 16; ++i) b[i] = (_Float16)p[i];
  return b;
}

__device__ __forceinline__ v16h load_a_frag_f16(const _Float16* A, int lda, int row,
                                                int k0, int half) {
  const _Float16* p = A + (long)row * lda + k0 + half * 16;
  v8h lo = *(const v8h*)p;
  v8h hi = *(const v8h*)(p + 8);
  return __builtin_shufflevector(lo, hi, 0, 1, 2, 3, 4, 5, 6, 7,
                                 8, 9, 10, 11, 12, 13, 14, 15);
}

// LDS byte offset of a shared-memory pointer: flat LDS aperture addresses map
// to LDS via addr[31:0] (ISA 10.2), so truncation yields the DS offset.
__device__ __forceinline__ unsigned lds_off(const void* p) {
  return (unsigned)(unsigned long long)p;
}

// ---------------------------------------------------------------------------
// Generic WMMA GEMM: C[M,N] = act(A[M,K] @ B[N,K]^T [+bias]) [+C].
// One wave per 16x64 strip; compile-time epilogue.
// ---------------------------------------------------------------------------
template <bool RELU, bool ACCUM, bool BIAS>
__global__ __launch_bounds__(128) void gemm16(const float* __restrict__ A, int lda,
                                              const float* __restrict__ Bw, int ldb,
                                              const float* __restrict__ bias,
                                              float* __restrict__ C, int ldc,
                                              int M, int N, int K) {
  int wave = threadIdx.x >> 5;
  int tile = blockIdx.x * (blockDim.x >> 5) + wave;
  int tilesN = N >> 6;
  int total = (M >> 4) * tilesN;
  if (tile >= total) return;                // wave-uniform: EXEC stays all-1s
  int tm = (tile / tilesN) << 4;
  int tn = (tile % tilesN) << 6;
  int lane = threadIdx.x & 31;
  int l16 = lane & 15, half = lane >> 4;

  v8f acc[4];
#pragma unroll
  for (int s = 0; s < 4; ++s) { v8f z = {}; acc[s] = z; }

  for (int k0 = 0; k0 < K; k0 += 32) {
    v16h a = load_a_frag_f32(A, lda, tm + l16, k0, half);
#pragma unroll
    for (int s = 0; s < 4; ++s) {
      v16h b = load_b_frag_f32(Bw + (long)(tn + s * 16 + l16) * ldb, k0, half);
      acc[s] = __builtin_amdgcn_wmma_f32_16x16x32_f16(false, a, false, b,
                                                      (short)0, acc[s], false, false);
    }
  }
#pragma unroll
  for (int s = 0; s < 4; ++s) {
#pragma unroll
    for (int v = 0; v < 8; ++v) {
      int row = tm + half * 8 + v, col = tn + s * 16 + l16;
      float o = acc[s][v];
      if (BIAS) o += bias[col];
      if (RELU) o = fmaxf(o, 0.f);
      long idx = (long)row * ldc + col;
      if (ACCUM) o += C[idx];
      C[idx] = o;
    }
  }
}

// ---------------------------------------------------------------------------
// segment_max via integer atomicMax on bit patterns (inputs are ReLU >= 0,
// dest pre-zeroed; self-loops guarantee non-empty segments). Deterministic.
// ---------------------------------------------------------------------------
__global__ void segmax_kernel(const float* __restrict__ p, const int* __restrict__ src,
                              const int* __restrict__ dst, float* __restrict__ hn,
                              int E2, int F) {
  long idx = (long)blockIdx.x * blockDim.x + threadIdx.x;
  if (idx >= (long)E2 * F) return;
  int e = (int)(idx / F), f = (int)(idx % F);
  float v = p[(long)src[e] * F + f];
  atomicMax((int*)&hn[(long)dst[e] * F + f], __float_as_int(v));
}

__global__ void zero_kernel(float* p, int n) {
  int i = blockIdx.x * blockDim.x + threadIdx.x;
  if (i < n) p[i] = 0.f;
}

__global__ void f32_to_f16_kernel(const float* __restrict__ s,
                                  _Float16* __restrict__ d, int n) {
  int i = blockIdx.x * blockDim.x + threadIdx.x;
  if (i < n) d[i] = (_Float16)s[i];
}

// relu + row l2norm for 4096 x 128
__global__ void relu_l2norm_kernel(float* __restrict__ h) {
  int row = blockIdx.x, t = threadIdx.x;
  float v = fmaxf(h[(long)row * 128 + t], 0.f);
  __shared__ float red[128];
  red[t] = v * v;
  __syncthreads();
  for (int s = 64; s > 0; s >>= 1) {
    if (t < s) red[t] += red[t + s];
    __syncthreads();
  }
  float nrm = fmaxf(sqrtf(red[0]), 1e-12f);
  h[(long)row * 128 + t] = v / nrm;
}

// SMOTE nearest-neighbor interp: block per candidate row i.
__global__ void smote_kernel(float* __restrict__ x, const float* __restrict__ gaps) {
  __shared__ float ci[128];
  __shared__ float bd[128];
  __shared__ int   bj[128];
  int i = blockIdx.x, t = threadIdx.x;
  ci[t] = x[(long)i * 128 + t];
  __syncthreads();
  float best = 3.4e38f;
  int bestj = 0;
  for (int j = t; j < 1024; j += 128) {
    if (j == i) continue;
    const float* cj = x + (long)j * 128;
    float d = 0.f;
#pragma unroll 8
    for (int k = 0; k < 128; ++k) {
      float df = ci[k] - cj[k];
      d += df * df;
    }
    if (d < best || (d == best && j < bestj)) { best = d; bestj = j; }
  }
  bd[t] = best; bj[t] = bestj;
  __syncthreads();
  for (int s = 64; s > 0; s >>= 1) {
    if (t < s) {
      if (bd[t + s] < bd[t] || (bd[t + s] == bd[t] && bj[t + s] < bj[t])) {
        bd[t] = bd[t + s]; bj[t] = bj[t + s];
      }
    }
    __syncthreads();
  }
  int nn = bj[0];
  float cv = ci[t];
  float diff = x[(long)nn * 128 + t] - cv;
#pragma unroll
  for (int r = 0; r < 2; ++r) {
    float g = gaps[i * 2 + r];
    x[(long)(4096 + i * 2 + r) * 128 + t] = cv + g * diff;
  }
}

// x (6144x128 f32) -> xTh (128x6144 f16)
__global__ void transpose_h_kernel(const float* __restrict__ x,
                                   _Float16* __restrict__ xTh) {
  int idx = blockIdx.x * blockDim.x + threadIdx.x;
  if (idx >= 6144 * 128) return;
  int i = idx >> 7, k = idx & 127;
  xTh[(long)k * 6144 + i] = (_Float16)x[idx];
}

// count nonzeros of adj (deterministic integer atomics)
__global__ void count_nz_kernel(const float* __restrict__ adj, int* cnt) {
  __shared__ int sh[256];
  int t = threadIdx.x;
  long stride = (long)gridDim.x * blockDim.x;
  int c = 0;
  for (long i = (long)blockIdx.x * blockDim.x + t; i < 16777216L; i += stride)
    if (adj[i] != 0.f) c++;
  sh[t] = c;
  __syncthreads();
  for (int s = 128; s > 0; s >>= 1) {
    if (t < s) sh[t] += sh[t + s];
    __syncthreads();
  }
  if (t == 0) atomicAdd(cnt, sh[0]);
}

__global__ void negw_kernel(float* scal) {
  int cnt = ((const int*)scal)[1];
  scal[2] = (float)cnt / (16777216.f - (float)cnt);
}

// ---------------------------------------------------------------------------
// Fused adjacency-reconstruction kernel; pred_adj never materialized.
// One wave per 16-row tile, 4 waves/block. The xTh K-tile (128x32 f16 = 8KB,
// identical for all waves of the block) is staged into LDS with async
// global->LDS copies, double buffered across J iterations (ASYNCcnt).
// ---------------------------------------------------------------------------
__global__ __launch_bounds__(128) void fused_adj_kernel(
    const _Float16* __restrict__ outh,  // 6144 x 128 f16
    const _Float16* __restrict__ xTh,   // 128 x 6144 f16
    const float* __restrict__ adj,      // 4096 x 4096
    const float* __restrict__ scal,     // [2] = neg_w
    float* __restrict__ neigh,          // 6144 x 128
    float* __restrict__ lossArr) {      // 384 per-tile partials
  __shared__ alignas(32) _Float16 tl[4][16][32];
  __shared__ alignas(32) _Float16 ldsX[2][128][32];   // double-buffered xTh tile
  int tid = threadIdx.x;
  int wave = tid >> 5;
  int tileI = blockIdx.x * 4 + wave;    // grid 96 x 4 waves = 384 exactly
  int lane = tid & 31;
  int l16 = lane & 15, half = lane >> 4;
  int i0 = tileI << 4;
  float neg_w = scal[2];
  bool rowInAdj = (i0 < 4096);          // exact: tiles are 16-aligned

  // async stage: thread tid copies 4 x 16B chunks of the 128x32 f16 tile
  unsigned ldsbase[2] = { lds_off(&ldsX[0][0][0]), lds_off(&ldsX[1][0][0]) };
  auto issue_xtile = [&](int buf, int J) {
#pragma unroll
    for (int r = 0; r < 4; ++r) {
      int idx = tid + 128 * r;          // 512 chunks: 128 rows x 4 k-chunks
      int row = idx >> 2, kc = idx & 3;
      unsigned ldso = ldsbase[buf] + (unsigned)(row * 64 + kc * 16);
      unsigned long long g =
          (unsigned long long)(const void*)(xTh + (long)row * 6144 + J + kc * 8);
      asm volatile("global_load_async_to_lds_b128 %0, %1, off"
                   :: "v"(ldso), "v"(g) : "memory");
    }
  };

  v16h aI[4];
#pragma unroll
  for (int t = 0; t < 4; ++t) aI[t] = load_a_frag_f16(outh, 128, i0 + l16, t * 32, half);

  v8f nacc[8];
#pragma unroll
  for (int c = 0; c < 8; ++c) { v8f z = {}; nacc[c] = z; }
  float rs[8] = {0, 0, 0, 0, 0, 0, 0, 0};
  float lossAcc = 0.f;

  const float* adjRow = adj + (long)(i0 + half * 8) * 4096;

  issue_xtile(0, 0);                    // prologue fill of buffer 0

  for (int J = 0; J < 6144; J += 32) {
    int buf = (J >> 5) & 1;
    bool more = (J + 32 < 6144);
    if (more) issue_xtile(buf ^ 1, J + 32);   // overlap next tile with compute

    if (rowInAdj && J + 32 < 4096)
      __builtin_prefetch(adjRow + J + 32 + l16, 0, 1);

    // ---- S = sigmoid(out_I @ out_J^T): preload all 8 B-frags, then two
    //      independent 4-WMMA chains (loads stay in flight across WMMAs) ----
    v16h bf[8];
#pragma unroll
    for (int jj2 = 0; jj2 < 2; ++jj2)
#pragma unroll
      for (int t = 0; t < 4; ++t)
        bf[jj2 * 4 + t] = *(const v16h*)(outh + (long)(J + jj2 * 16 + l16) * 128 +
                                         t * 32 + half * 16);
    v8f s0 = {}, s1 = {};
#pragma unroll
    for (int t = 0; t < 4; ++t) {
      s0 = __builtin_amdgcn_wmma_f32_16x16x32_f16(false, aI[t], false, bf[t],
                                                  (short)0, s0, false, false);
      s1 = __builtin_amdgcn_wmma_f32_16x16x32_f16(false, aI[t], false, bf[4 + t],
                                                  (short)0, s1, false, false);
    }
#pragma unroll
    for (int jj2 = 0; jj2 < 2; ++jj2) {
      const v8f& s = jj2 ? s1 : s0;
      int jj = jj2 * 16;
      bool doAdj = rowInAdj && (J + jj < 4096);   // wave-uniform guard
#pragma unroll
      for (int v = 0; v < 8; ++v) {
        float sv = 1.f / (1.f + __expf(-s[v]));
        if (doAdj) {
          float a = adjRow[(long)v * 4096 + J + jj + l16];
          float w = (a == 0.f) ? neg_w : 1.f;
          float d = sv - a;
          lossAcc += w * d * d;
        }
        float tv = (sv >= 0.5f) ? sv : 0.f;
        rs[v] += tv;
        tl[wave][half * 8 + v][jj + l16] = (_Float16)tv;
      }
    }

    // retire current buffer's async loads (in-order: wait past the 4 newer)
    if (more) asm volatile("s_wait_asynccnt 0x4" ::: "memory");
    else      asm volatile("s_wait_asynccnt 0x0" ::: "memory");
    __syncthreads();   // uniform: other waves' async writes + tl stores visible

    // ---- neigh_acc += T(16x32) @ x(32x128), B-frags from the LDS tile ----
    v8h lo = *(const v8h*)&tl[wave][l16][half * 16];
    v8h hi = *(const v8h*)&tl[wave][l16][half * 16 + 8];
    v16h a2 = __builtin_shufflevector(lo, hi, 0, 1, 2, 3, 4, 5, 6, 7,
                                      8, 9, 10, 11, 12, 13, 14, 15);
#pragma unroll
    for (int c = 0; c < 8; ++c) {
      v16h b = *(const v16h*)&ldsX[buf][c * 16 + l16][half * 16];
      nacc[c] = __builtin_amdgcn_wmma_f32_16x16x32_f16(false, a2, false, b,
                                                       (short)0, nacc[c], false, false);
    }
    __syncthreads();   // protect tl and ldsX[buf] before next overwrite
  }

  // rowsum: reduce across the 16 lanes of each half-group
#pragma unroll
  for (int v = 0; v < 8; ++v) {
    float r = rs[v];
    for (int o = 1; o < 16; o <<= 1) r += __shfl_xor(r, o, 16);
    rs[v] = r;
  }
#pragma unroll
  for (int c = 0; c < 8; ++c)
#pragma unroll
    for (int v = 0; v < 8; ++v)
      neigh[(long)(i0 + half * 8 + v) * 128 + c * 16 + l16] =
          nacc[c][v] / (rs[v] + 1.f);

  for (int o = 1; o < 32; o <<= 1) lossAcc += __shfl_xor(lossAcc, o, 32);
  if (lane == 0) lossArr[tileI] = lossAcc;  // deterministic (fixed-order final sum)
}

// logits (N=2), labels, loss -> d_out
__global__ void finalize_kernel(const float* __restrict__ hc,
                                const float* __restrict__ Wclf,
                                const float* __restrict__ bclf,
                                const int* __restrict__ labels,
                                const float* __restrict__ lossArr,
                                float* __restrict__ outF, int* __restrict__ outI) {
  int i = blockIdx.x * blockDim.x + threadIdx.x;
  if (i >= 6144) return;
  const float* hr = hc + (long)i * 128;
  float s0 = bclf[0], s1 = bclf[1];
#pragma unroll 8
  for (int k = 0; k < 128; ++k) {
    float v = hr[k];
    s0 += v * Wclf[k];
    s1 += v * Wclf[128 + k];
  }
  outF[2 * i] = s0;
  outF[2 * i + 1] = s1;
  outI[12288 + i] = (i < 4096) ? labels[i] : 1;
  if (i == 0) {
    float L = 0.f;
    for (int t = 0; t < 384; ++t) L += lossArr[t];
    outF[18432] = L;
  }
}

// ---------------------------------------------------------------------------
extern "C" void kernel_launch(void* const* d_in, const int* in_sizes, int n_in,
                              void* d_out, int out_size, void* d_ws, size_t ws_size,
                              hipStream_t stream) {
  (void)in_sizes; (void)n_in; (void)out_size; (void)ws_size;
  const float* feat     = (const float*)d_in[0];
  const float* adj      = (const float*)d_in[1];
  const int*   src      = (const int*)  d_in[2];
  const int*   dst      = (const int*)  d_in[3];
  const int*   labels   = (const int*)  d_in[4];
  const float* W_pool0  = (const float*)d_in[5];
  const float* b_pool0  = (const float*)d_in[6];
  const float* W_self0  = (const float*)d_in[7];
  const float* W_neigh0 = (const float*)d_in[8];
  const float* b0       = (const float*)d_in[9];
  const float* W_pool1  = (const float*)d_in[10];
  const float* b_pool1  = (const float*)d_in[11];
  const float* W_self1  = (const float*)d_in[12];
  const float* W_neigh1 = (const float*)d_in[13];
  const float* b1       = (const float*)d_in[14];
  const float* de_w     = (const float*)d_in[15];
  const float* W_conv   = (const float*)d_in[16];
  const float* W_clf    = (const float*)d_in[17];
  const float* b_clf    = (const float*)d_in[18];
  const float* gaps     = (const float*)d_in[19];

  float* ws = (float*)d_ws;
  float*     p0    = ws;                              // [0, 1048576)
  float*     hn0   = ws + 1048576;                    // [1048576, 2097152)
  float*     outb  = ws;                              //   reuse A after layer0
  _Float16*  outh  = (_Float16*)(ws + 786432);        //   reuse A
  _Float16*  xTh   = (_Float16*)(ws + 1179648);       //   reuse A
  float*     h     = ws + 2097152;                    // [2097152, 2621440)
  float*     p1    = ws + 2621440;                    // [2621440, 3145728)
  float*     hn1   = ws + 3145728;                    // [3145728, 3670016)
  float*     neigh = ws + 2621440;                    //   reuse C after layer1
  float*     x     = ws + 3670016;                    // [3670016, 4456448)
  float*     hc    = ws + 4456448;                    // [4456448, 5242880)
  float*     scal  = ws + 5242880;                    // [1]=cnt [2]=neg_w ; +16: loss partials

  zero_kernel<<<4096, 256, 0, stream>>>(hn0, 1048576);
  zero_kernel<<<2048, 256, 0, stream>>>(hn1, 524288);
  zero_kernel<<<2,    256, 0, stream>>>(scal, 512);

  // ---- SAGE layer 0 ----
  gemm16<true, false, true><<<256, 128, 0, stream>>>(feat, 256, W_pool0, 256, b_pool0,
                                                     p0, 256, 4096, 256, 256);
  segmax_kernel<<<(69632 * 256) / 256, 256, 0, stream>>>(p0, src, dst, hn0, 69632, 256);
  gemm16<false, false, true><<<128, 128, 0, stream>>>(feat, 256, W_self0, 256, b0,
                                                      h, 128, 4096, 128, 256);
  gemm16<false, true, false><<<128, 128, 0, stream>>>(hn0, 256, W_neigh0, 256, nullptr,
                                                      h, 128, 4096, 128, 256);
  relu_l2norm_kernel<<<4096, 128, 0, stream>>>(h);

  // ---- SAGE layer 1 ----
  gemm16<true, false, true><<<128, 128, 0, stream>>>(h, 128, W_pool1, 128, b_pool1,
                                                     p1, 128, 4096, 128, 128);
  segmax_kernel<<<(69632 * 128) / 256, 256, 0, stream>>>(p1, src, dst, hn1, 69632, 128);
  gemm16<false, false, true><<<128, 128, 0, stream>>>(h, 128, W_self1, 128, b1,
                                                      x, 128, 4096, 128, 128);
  gemm16<false, true, false><<<128, 128, 0, stream>>>(hn1, 128, W_neigh1, 128, nullptr,
                                                      x, 128, 4096, 128, 128);

  // ---- SMOTE synth rows, f16 side buffers, decoder GEMM ----
  smote_kernel<<<1024, 128, 0, stream>>>(x, gaps);
  transpose_h_kernel<<<3072, 256, 0, stream>>>(x, xTh);
  gemm16<false, false, false><<<192, 128, 0, stream>>>(x, 128, de_w, 128, nullptr,
                                                       outb, 128, 6144, 128, 128);
  f32_to_f16_kernel<<<3072, 256, 0, stream>>>(outb, outh, 786432);

  // ---- edge weighting + fused pred_adj (loss + neigh) ----
  count_nz_kernel<<<8192, 256, 0, stream>>>(adj, (int*)scal + 1);
  negw_kernel<<<1, 1, 0, stream>>>(scal);
  fused_adj_kernel<<<96, 128, 0, stream>>>(outh, xTh, adj, scal, neigh, scal + 16);

  // ---- hc = [x | neigh] @ W_conv^T  (split K) ----
  gemm16<false, false, false><<<192, 128, 0, stream>>>(x, 128, W_conv, 256, nullptr,
                                                       hc, 128, 6144, 128, 128);
  gemm16<false, true, false><<<192, 128, 0, stream>>>(neigh, 128, W_conv + 128, 256,
                                                      nullptr, hc, 128, 6144, 128, 128);

  // ---- logits + y + loss ----
  finalize_kernel<<<24, 256, 0, stream>>>(hc, W_clf, b_clf, labels, scal + 16,
                                          (float*)d_out, (int*)d_out);
}
